// dividedSpaceTimeAttention_88270167867454
// MI455X (gfx1250) — matrix operand order repaired and checked by reading, not compiled, not run on hardware
//
#include <hip/hip_runtime.h>
#include <math.h>

// ---------------- problem constants ----------------
#define HH     12
#define NP     196
#define FFRAME 8
#define DIMX   768
#define D3X    2304          // 3*DIM (projection width) and H*D3 merged width
#define DH3    192           // per-head slice width in merged space (D3)
#define BB     8
#define SS     1569          // 1 + 196*8
#define PU     195           // unique patches per frame (reference drops patch 0)
#define MXROWS (BB*SS)       // 12552
#define MUROWS (BB*PU)       // 1560
#define INV_SCALE (1.0f/96.0f)   // SCALE = sqrt(64)*12 = 96

// ---------------- types for WMMA ----------------
typedef __bf16 v16bf __attribute__((ext_vector_type(16)));
typedef float  v8f   __attribute__((ext_vector_type(8)));
typedef unsigned int u32x4 __attribute__((ext_vector_type(4)));

union Frag { v16bf v; u32x4 q[2]; };

__device__ __forceinline__ v8f wmma_bf16(v16bf a, v16bf b, v8f c) {
  return __builtin_amdgcn_wmma_f32_16x16x32_bf16(false, a, false, b, (short)0, c, false, false);
}

// ---------------- bf16 helpers (bit-level, RNE) ----------------
__device__ __forceinline__ unsigned short f2bf(float f) {
  unsigned int u = __float_as_uint(f);
  u += 0x7FFFu + ((u >> 16) & 1u);
  return (unsigned short)(u >> 16);
}
__device__ __forceinline__ float bf2f(unsigned short h) {
  return __uint_as_float(((unsigned int)h) << 16);
}

// ---------------- conversion kernels ----------------
// f32 -> (bf16 hi, bf16 lo) elementwise, row-major preserved
__global__ void __launch_bounds__(256)
k_cvt_pair(const float* __restrict__ src, unsigned short* __restrict__ hi,
           unsigned short* __restrict__ lo, long n) {
  long i = (long)blockIdx.x * 256 + threadIdx.x;
  if (i >= n) return;
  float f = src[i];
  unsigned short h = f2bf(f);
  hi[i] = h;
  lo[i] = f2bf(f - bf2f(h));
}

// f32 (K x N) -> bf16 hi/lo transposed (N x K)
__global__ void __launch_bounds__(256)
k_cvt_T_pair(const float* __restrict__ src, unsigned short* __restrict__ hiT,
             unsigned short* __restrict__ loT, int K, int N) {
  long i = (long)blockIdx.x * 256 + threadIdx.x;
  if (i >= (long)K * N) return;
  int k = (int)(i / N), n = (int)(i % N);
  float f = src[i];
  unsigned short h = f2bf(f);
  long o = (long)n * K + k;
  hiT[o] = h;
  loT[o] = f2bf(f - bf2f(h));
}

// ---------------- bf16x3 WMMA GEMM ----------------
// C[M,N] = A[M,K] * B[K,N] + bias, A given as bf16 hi/lo row-major,
// B given as bf16 hi/lo TRANSPOSED (N,K). Each wave computes a 16x64 tile.
// N % 64 == 0, K % 32 == 0; M arbitrary (load-clamp + store-guard).
__global__ void __launch_bounds__(256)
k_gemm_bf16x3(const unsigned short* __restrict__ Ahi,
              const unsigned short* __restrict__ Alo,
              const unsigned short* __restrict__ BThi,
              const unsigned short* __restrict__ BTlo,
              const float* __restrict__ bias,
              float* __restrict__ C,
              int M, int N, int K) {
  const int lane   = threadIdx.x & 31;
  const int tilesN = N >> 6;
  const int tilesM = (M + 15) >> 4;
  const long wid = (long)blockIdx.x * 8 + (threadIdx.x >> 5);   // wave-uniform
  if (wid >= (long)tilesM * tilesN) return;                     // whole wave exits: EXEC stays all-1 for WMMA
  const int tm  = (int)(wid / tilesN);
  const int tn  = (int)(wid % tilesN);
  const int hf  = lane >> 4;    // lane half (0: lanes0-15, 1: lanes16-31)
  const int l15 = lane & 15;

  int arow = tm * 16 + l15;
  if (arow > M - 1) arow = M - 1;   // clamped loads; stores are guarded
  const size_t aoff = (size_t)arow * K;

  v8f acc[4];
#pragma unroll
  for (int s = 0; s < 4; ++s)
#pragma unroll
    for (int r = 0; r < 8; ++r) acc[s][r] = 0.0f;

  for (int k0 = 0; k0 < K; k0 += 32) {
    // A fragment: this code's k order (same as B): element e -> k = hf*16 + e
    Frag ahi, alo;
    {
      const unsigned short* pa = Ahi + aoff + k0 + hf * 16;
      ahi.q[0] = *(const u32x4*)(pa);
      ahi.q[1] = *(const u32x4*)(pa + 8);
      const unsigned short* pl = Alo + aoff + k0 + hf * 16;
      alo.q[0] = *(const u32x4*)(pl);
      alo.q[1] = *(const u32x4*)(pl + 8);
    }
#pragma unroll
    for (int s = 0; s < 4; ++s) {
      // B fragment: this code's k order (same as A): element e -> k = hf*16 + e, n = l15
      Frag bhi, blo;
      const size_t boff = (size_t)(tn * 64 + s * 16 + l15) * K + k0 + hf * 16;
      bhi.q[0] = *(const u32x4*)(BThi + boff);
      bhi.q[1] = *(const u32x4*)(BThi + boff + 8);
      blo.q[0] = *(const u32x4*)(BTlo + boff);
      blo.q[1] = *(const u32x4*)(BTlo + boff + 8);
      acc[s] = wmma_bf16(ahi.v, bhi.v, acc[s]);   // hi*hi
      acc[s] = wmma_bf16(ahi.v, blo.v, acc[s]);   // hi*lo
      acc[s] = wmma_bf16(alo.v, bhi.v, acc[s]);   // lo*hi
    }
  }

#pragma unroll
  for (int s = 0; s < 4; ++s) {
    const int col = tn * 64 + s * 16 + l15;
    const float bvv = bias ? bias[col] : 0.0f;
#pragma unroll
    for (int r = 0; r < 8; ++r) {
      const int m = tm * 16 + hf * 8 + r;   // C layout: VGPR r -> M=r (+8 for lanes16-31)
      if (m < M) C[(size_t)m * N + col] = acc[s][r] + bvv;
    }
  }
}

// ---------------- block reductions ----------------
__device__ __forceinline__ float blk_red_max(float v, float* red) {
  int tid = threadIdx.x;
  red[tid] = v; __syncthreads();
  for (int s = blockDim.x >> 1; s > 0; s >>= 1) {
    if (tid < s) red[tid] = fmaxf(red[tid], red[tid + s]);
    __syncthreads();
  }
  float r = red[0]; __syncthreads();
  return r;
}
__device__ __forceinline__ float blk_red_sum(float v, float* red) {
  int tid = threadIdx.x;
  red[tid] = v; __syncthreads();
  for (int s = blockDim.x >> 1; s > 0; s >>= 1) {
    if (tid < s) red[tid] += red[tid + s];
    __syncthreads();
  }
  float r = red[0]; __syncthreads();
  return r;
}

// ---------------- CLS attention: q=Qx[row0], keys=Kx, values=Vx over 1569 ----------------
__global__ void __launch_bounds__(256)
k_cls_attn(const float* __restrict__ Qx, const float* __restrict__ Kx,
           const float* __restrict__ Vx, float* __restrict__ icls) {
  const int bh = blockIdx.x;         // b*12 + h
  const int b = bh / HH, h = bh % HH;
  const int tid = threadIdx.x;
  __shared__ float qsh[DH3];
  __shared__ float sc[SS];
  __shared__ float red[256];
  const size_t base = ((size_t)b * SS) * D3X + (size_t)h * DH3;
  if (tid < DH3) qsh[tid] = Qx[base + tid];
  __syncthreads();

  float lmax = -1e30f;
  for (int t = tid; t < SS; t += 256) {
    const float* kr = Kx + base + (size_t)t * D3X;
    float a = 0.0f;
    for (int d = 0; d < DH3; ++d) a += qsh[d] * kr[d];
    a *= INV_SCALE;
    sc[t] = a;
    lmax = fmaxf(lmax, a);
  }
  const float m = blk_red_max(lmax, red);
  float lsum = 0.0f;
  for (int t = tid; t < SS; t += 256) {
    float e = expf(sc[t] - m);
    sc[t] = e;
    lsum += e;
  }
  const float inv = 1.0f / blk_red_sum(lsum, red);
  __syncthreads();
  if (tid < DH3) {
    float acc = 0.0f;
    for (int t = 0; t < SS; ++t) acc += sc[t] * Vx[base + (size_t)t * D3X + tid];
    icls[(size_t)bh * DH3 + tid] = acc * inv;
  }
}

// ---------------- frame attention (8x8 per (b,h,p)); keys=Vx, values=Kx (ref name swap) ----------
__global__ void __launch_bounds__(64)
k_frame_attn(const float* __restrict__ Qx, const float* __restrict__ Kx,
             const float* __restrict__ Vx, float* __restrict__ t1u) {
  const int id = blockIdx.x;               // ((b*12+h)*195 + (p-1))
  const int pm1 = id % PU;
  const int bh  = id / PU;
  const int h = bh % HH, b = bh / HH;
  const int p = pm1 + 1;
  const int tid = threadIdx.x;
  __shared__ float qS[FFRAME][DH3], kS[FFRAME][DH3], vS[FFRAME][DH3];
  __shared__ float sS[64], pS[64], wS[FFRAME];

  for (int idx = tid; idx < FFRAME * DH3; idx += 64) {
    int f = idx / DH3, d = idx % DH3;
    size_t row = (size_t)b * SS + 1 + (size_t)f * NP + p;
    size_t off = row * D3X + (size_t)h * DH3 + d;
    qS[f][d] = Qx[off];
    kS[f][d] = Vx[off];   // k5 comes from the Wv projection
    vS[f][d] = Kx[off];   // v5 comes from the Wk projection
  }
  __syncthreads();

  const int f = tid >> 3, g = tid & 7;
  float a = 0.0f;
  for (int d = 0; d < DH3; ++d) a += qS[f][d] * kS[g][d];
  sS[tid] = a * INV_SCALE;
  __syncthreads();

  float m = sS[f * 8];
  for (int gg = 1; gg < 8; ++gg) m = fmaxf(m, sS[f * 8 + gg]);
  float e = expf(sS[tid] - m);
  pS[tid] = e;
  __syncthreads();
  float rs = 0.0f;
  for (int gg = 0; gg < 8; ++gg) rs += pS[f * 8 + gg];
  float P = e / rs;
  __syncthreads();
  pS[tid] = P;
  __syncthreads();
  if (tid < 8) {
    float w = 0.0f;
    for (int ff = 0; ff < 8; ++ff) w += pS[ff * 8 + tid];
    wS[tid] = w;
  }
  __syncthreads();
  for (int d = tid; d < DH3; d += 64) {
    float o = 0.0f;
    for (int g2 = 0; g2 < 8; ++g2) o += wS[g2] * vS[g2][d];
    t1u[((size_t)(b * PU + pm1)) * D3X + (size_t)h * DH3 + d] = o;
  }
}

// ---------------- round-2 attention, out[0]: 196x196 (rows mod 195) -------------------
// Q2 = queries (tiu@Wq), K2 = keys (tiu@Wv), V2 = values (tiu@Wk)
__global__ void __launch_bounds__(256)
k_attn2_self(const float* __restrict__ Q2, const float* __restrict__ K2,
             const float* __restrict__ V2, float* __restrict__ t2) {
  const int bh = blockIdx.x;
  const int b = bh / HH, h = bh % HH;
  const int tid = threadIdx.x;
  __shared__ float W[NP];
  __shared__ float S[32 * NP];
  __shared__ float rowm[32], rowr[32];
  for (int i = tid; i < NP; i += 256) W[i] = 0.0f;
  __syncthreads();
  const size_t cbase = (size_t)h * DH3;

  for (int j0 = 0; j0 < NP; j0 += 32) {
    const int nj = (NP - j0 < 32) ? (NP - j0) : 32;
    for (int idx = tid; idx < nj * NP; idx += 256) {
      int jj = idx / NP, l = idx % NP;
      int uq = (j0 + jj) % PU, ul = l % PU;
      const float* qr = Q2 + ((size_t)(b * PU + uq)) * D3X + cbase;
      const float* kr = K2 + ((size_t)(b * PU + ul)) * D3X + cbase;
      float a = 0.0f;
      for (int d = 0; d < DH3; ++d) a += qr[d] * kr[d];
      S[jj * NP + l] = a * INV_SCALE;
    }
    __syncthreads();
    if (tid < nj) {
      float m = -1e30f;
      for (int l = 0; l < NP; ++l) m = fmaxf(m, S[tid * NP + l]);
      float r = 0.0f;
      for (int l = 0; l < NP; ++l) r += expf(S[tid * NP + l] - m);
      rowm[tid] = m;
      rowr[tid] = 1.0f / r;
    }
    __syncthreads();
    for (int idx = tid; idx < nj * NP; idx += 256) {
      int jj = idx / NP, l = idx % NP;
      atomicAdd(&W[l], expf(S[jj * NP + l] - rowm[jj]) * rowr[jj]);   // ds_add_f32
    }
    __syncthreads();
  }
  if (tid < DH3) {
    float o = 0.0f;
    for (int l = 0; l < NP; ++l) {
      int ul = l % PU;
      o += W[l] * V2[((size_t)(b * PU + ul)) * D3X + cbase + tid];
    }
    t2[((size_t)bh * FFRAME + 0) * DH3 + tid] = o;
  }
}

// ---------------- round-2 attention, out[1..7]: 196 queries x 8 keys -------------------
__global__ void __launch_bounds__(256)
k_attn2_cross(const float* __restrict__ Q2, const float* __restrict__ K2,
              const float* __restrict__ V2, float* __restrict__ t2) {
  const int id = blockIdx.x;        // (b*12+h)*7 + (xi-1)
  const int xi = (id % 7) + 1;
  const int bh = id / 7;
  const int b = bh / HH, h = bh % HH;
  const int tid = threadIdx.x;
  __shared__ float kS[FFRAME][DH3], vS[FFRAME][DH3];
  __shared__ float S[NP * FFRAME];
  __shared__ float W[FFRAME];
  if (tid < FFRAME) W[tid] = 0.0f;
  const size_t cbase = (size_t)h * DH3;
  for (int idx = tid; idx < FFRAME * DH3; idx += 256) {
    int g = idx / DH3, d = idx % DH3;
    int ug = (xi + g) % PU;
    size_t ro = ((size_t)(b * PU + ug)) * D3X + cbase + d;
    kS[g][d] = K2[ro];
    vS[g][d] = V2[ro];
  }
  __syncthreads();
  for (int idx = tid; idx < NP * FFRAME; idx += 256) {
    int j = idx >> 3, g = idx & 7;
    int uq = (xi + j) % PU;
    const float* qr = Q2 + ((size_t)(b * PU + uq)) * D3X + cbase;
    float a = 0.0f;
    for (int d = 0; d < DH3; ++d) a += qr[d] * kS[g][d];
    S[idx] = a * INV_SCALE;
  }
  __syncthreads();
  if (tid < NP) {
    float m = -1e30f;
    for (int g = 0; g < 8; ++g) m = fmaxf(m, S[tid * 8 + g]);
    float e[8];
    float r = 0.0f;
    for (int g = 0; g < 8; ++g) { e[g] = expf(S[tid * 8 + g] - m); r += e[g]; }
    float inv = 1.0f / r;
    for (int g = 0; g < 8; ++g) atomicAdd(&W[g], e[g] * inv);
  }
  __syncthreads();
  if (tid < DH3) {
    float o = 0.0f;
    for (int g = 0; g < 8; ++g) o += W[g] * vS[g][tid];
    t2[((size_t)bh * FFRAME + xi) * DH3 + tid] = o;
  }
}

// ---------------- assemble Ybig (12552 x 2304) as bf16 hi/lo for final GEMM ------------
__global__ void __launch_bounds__(256)
k_assemble(const float* __restrict__ icls, const float* __restrict__ t2,
           unsigned short* __restrict__ Yhi, unsigned short* __restrict__ Ylo, long total) {
  long i = (long)blockIdx.x * 256 + threadIdx.x;
  if (i >= total) return;
  long row = i / D3X;
  int c = (int)(i % D3X);
  int b = (int)(row / SS), s = (int)(row % SS);
  int h = c / DH3, d = c % DH3;
  float v;
  if (s == 0) v = icls[((size_t)(b * HH + h)) * DH3 + d];
  else        v = t2[(((size_t)(b * HH + h)) * FFRAME + ((s - 1) & 7)) * DH3 + d];
  unsigned short hi = f2bf(v);
  Yhi[i] = hi;
  Ylo[i] = f2bf(v - bf2f(hi));
}

// ---------------- host side ----------------
static inline int cdivl(long a, long b) { return (int)((a + b - 1) / b); }

static void launch_gemm(const unsigned short* Ahi, const unsigned short* Alo,
                        const unsigned short* BThi, const unsigned short* BTlo,
                        const float* bias, float* C, int M, int N, int K,
                        hipStream_t stream) {
  long waves = (long)((M + 15) / 16) * (N / 64);
  int blocks = (int)((waves + 7) / 8);
  k_gemm_bf16x3<<<blocks, 256, 0, stream>>>(Ahi, Alo, BThi, BTlo, bias, C, M, N, K);
}

extern "C" void kernel_launch(void* const* d_in, const int* in_sizes, int n_in,
                              void* d_out, int out_size, void* d_ws, size_t ws_size,
                              hipStream_t stream) {
  (void)in_sizes; (void)n_in; (void)out_size; (void)ws_size;
  const float* x  = (const float*)d_in[0];
  const float* Wq = (const float*)d_in[1];
  const float* bq = (const float*)d_in[2];
  const float* Wk = (const float*)d_in[3];
  const float* bk = (const float*)d_in[4];
  const float* Wv = (const float*)d_in[5];
  const float* bv = (const float*)d_in[6];
  const float* Wt = (const float*)d_in[7];
  const float* bt = (const float*)d_in[8];
  const float* Wf = (const float*)d_in[9];
  const float* bf = (const float*)d_in[10];
  float* out = (float*)d_out;

  char* ws = (char*)d_ws;
  size_t off = 0;
  auto take = [&](size_t bytes) -> char* {
    char* p = ws + off;
    off = (off + bytes + 255) & ~((size_t)255);
    return p;
  };

  const long nX  = (long)MXROWS * DIMX;   // 9.64M
  const long nW  = (long)DIMX * D3X;      // 1.77M (every weight has this count)
  const long nQ  = (long)MXROWS * D3X;    // 28.9M
  const long nT1 = (long)MUROWS * D3X;
  const long nTi = (long)MUROWS * DIMX;

  unsigned short* xhi  = (unsigned short*)take((size_t)nX * 2);
  unsigned short* xlo  = (unsigned short*)take((size_t)nX * 2);
  unsigned short* wqTh = (unsigned short*)take((size_t)nW * 2);
  unsigned short* wqTl = (unsigned short*)take((size_t)nW * 2);
  unsigned short* wkTh = (unsigned short*)take((size_t)nW * 2);
  unsigned short* wkTl = (unsigned short*)take((size_t)nW * 2);
  unsigned short* wvTh = (unsigned short*)take((size_t)nW * 2);
  unsigned short* wvTl = (unsigned short*)take((size_t)nW * 2);
  unsigned short* wtTh = (unsigned short*)take((size_t)nW * 2);
  unsigned short* wtTl = (unsigned short*)take((size_t)nW * 2);
  unsigned short* wfTh = (unsigned short*)take((size_t)nW * 2);
  unsigned short* wfTl = (unsigned short*)take((size_t)nW * 2);
  float* Qx   = (float*)take((size_t)nQ * 4);
  float* Kx   = (float*)take((size_t)nQ * 4);
  float* Vx   = (float*)take((size_t)nQ * 4);
  float* icls = (float*)take((size_t)BB * HH * DH3 * 4);
  float* t1u  = (float*)take((size_t)nT1 * 4);
  unsigned short* t1uh = (unsigned short*)take((size_t)nT1 * 2);
  unsigned short* t1ul = (unsigned short*)take((size_t)nT1 * 2);
  float* tiu  = (float*)take((size_t)nTi * 4);
  unsigned short* tiuh = (unsigned short*)take((size_t)nTi * 2);
  unsigned short* tiul = (unsigned short*)take((size_t)nTi * 2);
  float* A2q  = (float*)take((size_t)nT1 * 4);
  float* A2k  = (float*)take((size_t)nT1 * 4);
  float* A2v  = (float*)take((size_t)nT1 * 4);
  float* t2   = (float*)take((size_t)BB * HH * FFRAME * DH3 * 4);
  unsigned short* Yhi = (unsigned short*)take((size_t)nQ * 2);
  unsigned short* Ylo = (unsigned short*)take((size_t)nQ * 2);

  // 1) convert inputs: x row-major pair; weights transposed pair
  k_cvt_pair<<<cdivl(nX, 256), 256, 0, stream>>>(x, xhi, xlo, nX);
  k_cvt_T_pair<<<cdivl(nW, 256), 256, 0, stream>>>(Wq, wqTh, wqTl, DIMX, D3X);
  k_cvt_T_pair<<<cdivl(nW, 256), 256, 0, stream>>>(Wk, wkTh, wkTl, DIMX, D3X);
  k_cvt_T_pair<<<cdivl(nW, 256), 256, 0, stream>>>(Wv, wvTh, wvTl, DIMX, D3X);
  k_cvt_T_pair<<<cdivl(nW, 256), 256, 0, stream>>>(Wt, wtTh, wtTl, D3X, DIMX);
  k_cvt_T_pair<<<cdivl(nW, 256), 256, 0, stream>>>(Wf, wfTh, wfTl, D3X, DIMX);

  // 2) full projections over all tokens (covers patches + cls paths)
  launch_gemm(xhi, xlo, wqTh, wqTl, bq, Qx, MXROWS, D3X, DIMX, stream);
  launch_gemm(xhi, xlo, wkTh, wkTl, bk, Kx, MXROWS, D3X, DIMX, stream);
  launch_gemm(xhi, xlo, wvTh, wvTl, bv, Vx, MXROWS, D3X, DIMX, stream);

  // 3) cls attention + per-patch frame attention (unique rows only)
  k_cls_attn<<<BB * HH, 256, 0, stream>>>(Qx, Kx, Vx, icls);
  k_frame_attn<<<BB * HH * PU, 64, 0, stream>>>(Qx, Kx, Vx, t1u);

  // 4) ti = t1 @ Wt + bt (195 unique rows per batch)
  k_cvt_pair<<<cdivl(nT1, 256), 256, 0, stream>>>(t1u, t1uh, t1ul, nT1);
  launch_gemm(t1uh, t1ul, wtTh, wtTl, bt, tiu, MUROWS, DIMX, D3X, stream);

  // 5) round-2 projections (ref: q2<-Wq, v2<-Wk, k2<-Wv)
  k_cvt_pair<<<cdivl(nTi, 256), 256, 0, stream>>>(tiu, tiuh, tiul, nTi);
  launch_gemm(tiuh, tiul, wqTh, wqTl, bq, A2q, MUROWS, D3X, DIMX, stream);
  launch_gemm(tiuh, tiul, wkTh, wkTl, bk, A2k, MUROWS, D3X, DIMX, stream);
  launch_gemm(tiuh, tiul, wvTh, wvTl, bv, A2v, MUROWS, D3X, DIMX, stream);

  // 6) round-2 attentions -> t2 (keys = A2v, values = A2k)
  k_attn2_self<<<BB * HH, 256, 0, stream>>>(A2q, A2v, A2k, t2);
  k_attn2_cross<<<BB * HH * 7, 256, 0, stream>>>(A2q, A2v, A2k, t2);

  // 7) assemble merged sequence and final projection @ Wf + bf -> out
  k_assemble<<<cdivl(nQ, 256), 256, 0, stream>>>(icls, t2, Yhi, Ylo, nQ);
  launch_gemm(Yhi, Ylo, wfTh, wfTl, bf, out, MXROWS, DIMX, D3X, stream);
}
